// WindowAttention_44117904064565
// MI455X (gfx1250) — compile-verified, matrix operand order checked by reading, not run on hardware
//
#include <hip/hip_runtime.h>
#include <hip/hip_bf16.h>

// ---------------------------------------------------------------------------
// Swin window attention, fused WMMA pipeline for gfx1250 (wave32).
//   K0: convert+transpose weights to f16 ([N x K] row-major) in ws
//   K1: per-window fused QKV GEMM + attention -> attn_out f16 (ws)
//   K2: projection GEMM + bias + window-reverse scatter -> f32 out
// ---------------------------------------------------------------------------

typedef __attribute__((ext_vector_type(16))) _Float16 v16h;
typedef __attribute__((ext_vector_type(8)))  _Float16 v8h;
typedef __attribute__((ext_vector_type(8)))  float    v8f;

#define WMMA_F16(A, B, C) \
  __builtin_amdgcn_wmma_f32_16x16x32_f16(false, (A), false, (B), (short)0, (C), false, false)

// Load one 16-lane-striped f16 fragment: per the CDNA5 16-bit A/B layout a
// lane's 16 halves are two contiguous 8-half (16B) chunks at +0 and +16
// relative to k0 + 8*(lane>=16). Works for LDS (ds_load_b128) and global
// (global_load_b128) pointers.
__device__ __forceinline__ v16h frag2(const _Float16* p) {
  v8h lo = *(const v8h*)(p);
  v8h hi = *(const v8h*)(p + 16);
  v16h r;
#pragma unroll
  for (int i = 0; i < 8; ++i) { r[i] = lo[i]; r[i + 8] = hi[i]; }
  return r;
}

__device__ __forceinline__ int rel_idx(int i, int j) {
  int dh = i / 7 - j / 7 + 6;   // [0,12]
  int dw = i % 7 - j % 7 + 6;   // [0,12]
  return dh * 13 + dw;          // [0,168]
}

// ---------------- K0: weight convert + transpose -> f16 [N x K] -------------
__global__ __launch_bounds__(256) void k_cvt(const float* __restrict__ qkv_w,
                                             const float* __restrict__ proj_w,
                                             _Float16* __restrict__ wqkvT,
                                             _Float16* __restrict__ wprojT) {
  int idx = blockIdx.x * 256 + threadIdx.x;
  if (idx < 384 * 1152) {
    int r = idx / 1152, c = idx % 1152;              // qkv_w is [K=384][N=1152]
    wqkvT[(size_t)c * 384 + r] = (_Float16)qkv_w[idx];
  } else {
    int i2 = idx - 384 * 1152;
    if (i2 < 384 * 384) {
      int r = i2 / 384, c = i2 % 384;                // proj_w is [K=384][N=384]
      wprojT[(size_t)c * 384 + r] = (_Float16)proj_w[i2];
    }
  }
}

// ---------------- K1: fused per-window QKV + attention ----------------------
// LDS layout (bytes):
//   [0,       76048)  qkvs: 49 x 776 f16  (q cols 0..383 pre-scaled, k cols 384..767)
//   [76048,  125200)  vts : 384 x 64 f16  (V transposed: row=channel, col=token)
//   [125200, 198928)  phase-shared: xs (49 x 392 f16) then per-wave scores
//                     swv: 8 waves x (64 x 72) f16
#define LDQ   776
#define LDV   64
#define LDX   392
#define LDS_S 72
#define SMEM1 198928
#define QSCALE 0.17677669529663687f   // 1/sqrt(32)

__global__ __launch_bounds__(256) void k_qkv_attn(const float* __restrict__ x,
                                                  const float* __restrict__ qkv_b,
                                                  const _Float16* __restrict__ wqkvT,
                                                  const float* __restrict__ rpb,
                                                  _Float16* __restrict__ attn_out) {
  extern __shared__ __align__(16) char smem[];
  _Float16* qkvs = (_Float16*)smem;
  _Float16* vts  = (_Float16*)(smem + 76048);
  _Float16* xs   = (_Float16*)(smem + 125200);

  const int tid  = threadIdx.x;
  const int lane = tid & 31;
  const int wid  = tid >> 5;          // 8 waves
  const int lh   = lane >> 4;         // half-wave select
  const int ln   = lane & 15;
  const int kb   = lh << 3;           // +8 halves for upper half-wave

  const int w  = blockIdx.x;          // window id 0..2047
  const int b  = w >> 6;
  const int wh = (w >> 3) & 7;
  const int ww = w & 7;

  // ---- stage x window (f32 -> f16, 49 x 384) and zero vts ----
  for (int i = tid; i < 3072; i += 256) ((v8h*)vts)[i] = (v8h){};
  for (int i = tid; i < 49 * 96; i += 256) {
    int row = i / 96;
    int col = (i % 96) * 4;
    size_t gbase = (((size_t)(b * 56 + wh * 7 + row / 7)) * 56 + (ww * 7 + row % 7)) * 384 + col;
    float4 v = *(const float4*)(x + gbase);
    xs[row * LDX + col + 0] = (_Float16)v.x;
    xs[row * LDX + col + 1] = (_Float16)v.y;
    xs[row * LDX + col + 2] = (_Float16)v.z;
    xs[row * LDX + col + 3] = (_Float16)v.w;
  }
  __syncthreads();

  // ---- QKV GEMM: (64pad x 384) @ (384 x 1152); wave handles 9 col-tiles ----
#pragma unroll 1
  for (int t = 0; t < 9; ++t) {
    const int ncol0 = (wid + 8 * t) * 16;
    const int ncol  = ncol0 + ln;
    const float bn  = qkv_b[ncol];
    v8f a0{}, a1{}, a2{}, a3{};
#pragma unroll
    for (int ks = 0; ks < 12; ++ks) {
      const int k0 = ks * 32 + kb;
      v16h bf = frag2(wqkvT + (size_t)ncol * 384 + k0);
      v16h f0 = frag2(xs + (0 * 16 + ln) * LDX + k0);
      a0 = WMMA_F16(f0, bf, a0);
      v16h f1 = frag2(xs + (1 * 16 + ln) * LDX + k0);
      a1 = WMMA_F16(f1, bf, a1);
      v16h f2 = frag2(xs + (2 * 16 + ln) * LDX + k0);
      a2 = WMMA_F16(f2, bf, a2);
      v16h f3 = frag2(xs + (3 * 16 + ln) * LDX + k0);
      a3 = WMMA_F16(f3, bf, a3);
    }
    const v8f accs[4] = {a0, a1, a2, a3};
#pragma unroll
    for (int mt = 0; mt < 4; ++mt) {
#pragma unroll
      for (int r = 0; r < 8; ++r) {
        int m = mt * 16 + r + 8 * lh;
        if (m < 49) {
          float v = accs[mt][r] + bn;
          if (ncol < 384) v *= QSCALE;                    // pre-scale q
          if (ncol < 768) qkvs[m * LDQ + ncol] = (_Float16)v;
          else            vts[(ncol - 768) * LDV + m] = (_Float16)v;
        }
      }
    }
  }
  __syncthreads();

  // ---- attention: 12 heads over 8 waves (2 rounds) ----
  _Float16* swv = (_Float16*)(smem + 125200) + wid * 64 * LDS_S;
#pragma unroll 1
  for (int rr = 0; rr < 2; ++rr) {
    const int h = wid + rr * 8;
    if (h >= 12) continue;

    // S = q @ k^T  (49pad64 x 49pad64, K = hd = 32 -> one WMMA per tile)
    const int cq = h * 32 + kb;
    const int ck = 384 + h * 32 + kb;
    v16h kf[4];
#pragma unroll
    for (int nt = 0; nt < 4; ++nt) kf[nt] = frag2(qkvs + (nt * 16 + ln) * LDQ + ck);
    v8f s[4][4];
#pragma unroll
    for (int mt = 0; mt < 4; ++mt) {
      v16h qf = frag2(qkvs + (mt * 16 + ln) * LDQ + cq);
      v8f z{};
#pragma unroll
      for (int nt = 0; nt < 4; ++nt) s[mt][nt] = WMMA_F16(qf, kf[nt], z);
    }

    // bias + mask + softmax (rows live in one 16-lane half of the wave)
#pragma unroll
    for (int mt = 0; mt < 4; ++mt) {
#pragma unroll
      for (int r = 0; r < 8; ++r) {
        int m  = mt * 16 + r + 8 * lh;
        int mi = m < 49 ? m : 48;
#pragma unroll
        for (int nt = 0; nt < 4; ++nt) {
          int n = nt * 16 + ln;
          if (n < 49) s[mt][nt][r] += rpb[rel_idx(mi, n) * 12 + h];
          else        s[mt][nt][r] = -1e30f;              // overwrite: kills pad garbage
        }
        float mx = fmaxf(fmaxf(s[mt][0][r], s[mt][1][r]), fmaxf(s[mt][2][r], s[mt][3][r]));
        mx = fmaxf(mx, __shfl_xor(mx, 1));
        mx = fmaxf(mx, __shfl_xor(mx, 2));
        mx = fmaxf(mx, __shfl_xor(mx, 4));
        mx = fmaxf(mx, __shfl_xor(mx, 8));
        float sum = 0.f;
#pragma unroll
        for (int nt = 0; nt < 4; ++nt) {
          float e = __expf(s[mt][nt][r] - mx);
          s[mt][nt][r] = e;
          sum += e;
        }
        sum += __shfl_xor(sum, 1);
        sum += __shfl_xor(sum, 2);
        sum += __shfl_xor(sum, 4);
        sum += __shfl_xor(sum, 8);
        float inv = 1.0f / sum;
#pragma unroll
        for (int nt = 0; nt < 4; ++nt)
          swv[m * LDS_S + nt * 16 + ln] = (_Float16)(s[mt][nt][r] * inv);
      }
    }
    // make this wave's LDS score writes visible to all of its lanes
    asm volatile("s_wait_dscnt 0" ::: "memory");

    // O = P(49pad64 x 64) @ V(64 x 32): K padded cols of P are exactly 0
    v8f o[4][2];
#pragma unroll
    for (int mt = 0; mt < 4; ++mt)
#pragma unroll
      for (int nt = 0; nt < 2; ++nt) o[mt][nt] = (v8f){};
#pragma unroll
    for (int kt = 0; kt < 2; ++kt) {
      const int k0 = kt * 32 + kb;
      v16h vf[2];
#pragma unroll
      for (int nt = 0; nt < 2; ++nt)
        vf[nt] = frag2(vts + (h * 32 + nt * 16 + ln) * LDV + k0);
#pragma unroll
      for (int mt = 0; mt < 4; ++mt) {
        v16h pf = frag2(swv + (mt * 16 + ln) * LDS_S + k0);
#pragma unroll
        for (int nt = 0; nt < 2; ++nt) o[mt][nt] = WMMA_F16(pf, vf[nt], o[mt][nt]);
      }
    }
#pragma unroll
    for (int mt = 0; mt < 4; ++mt)
#pragma unroll
      for (int nt = 0; nt < 2; ++nt)
#pragma unroll
        for (int r = 0; r < 8; ++r) {
          int m = mt * 16 + r + 8 * lh;
          if (m < 49)
            attn_out[((size_t)w * 49 + m) * 384 + h * 32 + nt * 16 + ln] = (_Float16)o[mt][nt][r];
        }
  }
}

// ---------------- K2: projection GEMM + window reverse ----------------------
__global__ __launch_bounds__(256) void k_proj(const _Float16* __restrict__ attn,
                                              const _Float16* __restrict__ wprojT,
                                              const float* __restrict__ proj_b,
                                              float* __restrict__ out) {
  const int lane = threadIdx.x & 31;
  const int wid  = threadIdx.x >> 5;
  const int lh   = lane >> 4;
  const int ln   = lane & 15;
  const int kb   = lh << 3;
  const int mstrip = blockIdx.x * 128 + wid * 16;   // 784 * 128 = 100352 exactly
  const int nbase  = blockIdx.y * 64;

  v8f acc[4];
#pragma unroll
  for (int nt = 0; nt < 4; ++nt) acc[nt] = (v8f){};

  const _Float16* ap = attn + (size_t)(mstrip + ln) * 384 + kb;
#pragma unroll
  for (int ks = 0; ks < 12; ++ks) {
    v16h a = frag2(ap + ks * 32);
#pragma unroll
    for (int nt = 0; nt < 4; ++nt) {
      v16h bf = frag2(wprojT + (size_t)(nbase + nt * 16 + ln) * 384 + ks * 32 + kb);
      acc[nt] = WMMA_F16(a, bf, acc[nt]);
    }
  }

#pragma unroll
  for (int nt = 0; nt < 4; ++nt) {
    const int ncol = nbase + nt * 16 + ln;
    const float bias = proj_b[ncol];
#pragma unroll
    for (int r = 0; r < 8; ++r) {
      int m   = mstrip + r + 8 * lh;
      int w   = m / 49, tok = m % 49;
      int b   = w >> 6, wh = (w >> 3) & 7, ww = w & 7;
      int hh  = wh * 7 + tok / 7;
      int wc  = ww * 7 + tok % 7;
      out[(((size_t)b * 56 + hh) * 56 + wc) * 384 + ncol] = acc[nt][r] + bias;
    }
  }
}

// ---------------------------------------------------------------------------
extern "C" void kernel_launch(void* const* d_in, const int* in_sizes, int n_in,
                              void* d_out, int out_size, void* d_ws, size_t ws_size,
                              hipStream_t stream) {
  (void)in_sizes; (void)n_in; (void)out_size; (void)ws_size;
  const float* x      = (const float*)d_in[0];
  const float* qkv_w  = (const float*)d_in[1];
  const float* qkv_b  = (const float*)d_in[2];
  const float* proj_w = (const float*)d_in[3];
  const float* proj_b = (const float*)d_in[4];
  const float* rpb    = (const float*)d_in[5];
  float* out = (float*)d_out;

  // workspace carve (needs ~78.3 MB): wqkvT | wprojT | attn_out(f16)
  _Float16* wqkvT  = (_Float16*)d_ws;                                  // 884,736 B
  _Float16* wprojT = (_Float16*)((char*)d_ws + 884736);                // 294,912 B
  _Float16* attn   = (_Float16*)((char*)d_ws + 884736 + 294912);       // 77,070,336 B

  k_cvt<<<2304, 256, 0, stream>>>(qkv_w, proj_w, wqkvT, wprojT);
  k_qkv_attn<<<2048, 256, SMEM1, stream>>>(x, qkv_b, wqkvT, rpb, attn);
  k_proj<<<dim3(784, 6), 256, 0, stream>>>(attn, wprojT, proj_b, out);
}
